// encoderOnlyComplex_9328668967040
// MI455X (gfx1250) — matrix operand order repaired and checked by reading, not compiled, not run on hardware
//
#include <hip/hip_runtime.h>

// ---------------------------------------------------------------------------
// CDNA5 (gfx1250) implementation of the VoxelMorph-style registration net.
// Conv3d layers run as implicit-GEMM on v_wmma_f32_16x16x32_bf16 (bf16 A/B,
// f32 accumulate). One wave per 16-voxel tile; fragments are gathered
// directly from global memory into matching contiguous-k (K=16*half+e) wave32 WMMA lane layouts
// with branchless (clamped-address + select) bounds handling. All hot offset
// arithmetic is 32-bit so the backend can use saddr+voffset addressing.
// ---------------------------------------------------------------------------

typedef __attribute__((ext_vector_type(16))) __bf16 v16bf;
typedef __attribute__((ext_vector_type(8)))  float  v8f;

union BfFrag { unsigned short h[16]; v16bf v; };

__device__ __forceinline__ unsigned short f2bf(float f) {
  unsigned int u = __float_as_uint(f);
  unsigned int r = u + 0x7FFFu + ((u >> 16) & 1u);   // round-to-nearest-even
  return (unsigned short)(r >> 16);
}

__device__ __forceinline__ int iclamp(int v, int lo, int hi) {
  return v < lo ? lo : (v > hi ? hi : v);
}

// ---------------------------------------------------------------------------
// Implicit-GEMM 3x3x3 conv.  Y[co, vox] = sum_k W[co,k] * im2col(X)[k, vox].
// blockDim = 256 (8 independent waves); each wave computes one 16-voxel tile
// for up to 32 out channels (two 16x16 accumulators sharing one B fragment).
// K = Cin*27, chunked by 32 with zero padding.
//
// Fragment lane layouts (wave32):
//   A 16x32 bf16 : lane row=M (both halves); vgpr v -> K=16*half+2v+{0,1}
//   B 32x16 bf16 : lane row=N; element e -> K=16*half+e
//   C/D 16x16 f32: lane row=N; vgpr r -> M=r+8*half
// ---------------------------------------------------------------------------
__global__ void __launch_bounds__(256)
conv3d_wmma_kernel(const float* __restrict__ X, const float* __restrict__ Wg,
                   const float* __restrict__ Bi, float* __restrict__ Y,
                   int Cin, int Cout, int D, int H, int W, int relu)
{
  const int V    = D * H * W;
  const int K    = Cin * 27;
  const int lane = threadIdx.x & 31;
  const int row  = lane & 15;
  const int half = lane >> 4;
  const int tile = blockIdx.x * 8 + (threadIdx.x >> 5);
  const bool twoCo = (Cout > 16);          // uniform scalar branch

  // This lane's voxel column (fixed for the whole kernel).
  const int  vox   = tile * 16 + row;
  const bool voxOK = (vox < V);
  const int  voxc  = voxOK ? vox : (V - 1);
  const int  z = voxc / (H * W);
  const int  r0 = voxc - z * (H * W);
  const int  y = r0 / W;
  const int  x = r0 - y * W;

  const int coClamp0 = iclamp(row, 0, Cout - 1) * K;        // 32-bit row bases
  const int coClamp1 = iclamp(16 + row, 0, Cout - 1) * K;

  v8f acc0 = {};
  v8f acc1 = {};

  const int KT = (K + 31) >> 5;
  for (int kc = 0; kc < KT; ++kc) {
    const int k0 = kc * 32;

    // ---- B fragment: direct branchless im2col gather (16 loads/lane) ----
    BfFrag b;
    #pragma unroll
    for (int e = 0; e < 16; ++e) {
      int  gk  = k0 + 16 * half + e;
      bool kOK = (gk < K);
      int  gkc = kOK ? gk : 0;
      int  cin = gkc / 27;
      int  t   = gkc - cin * 27;
      int  kz  = t / 9;
      int  r2  = t - kz * 9;
      int  ky  = r2 / 3;
      int  kx  = r2 - ky * 3;
      int  zz = z + kz - 1, yy = y + ky - 1, xw = x + kx - 1;
      bool ok = kOK && voxOK &&
                ((unsigned)zz < (unsigned)D) &&
                ((unsigned)yy < (unsigned)H) &&
                ((unsigned)xw < (unsigned)W);
      int  zc = iclamp(zz, 0, D - 1), yc = iclamp(yy, 0, H - 1), xc = iclamp(xw, 0, W - 1);
      float v = X[cin * V + (zc * H + yc) * W + xc];   // always-valid 32-bit offset
      b.h[e] = f2bf(ok ? v : 0.f);
    }

    // ---- A fragment(s): direct branchless weight gather ----
    BfFrag a0, a1;
    #pragma unroll
    for (int vv = 0; vv < 8; ++vv) {
      int ka = 16 * half + 2 * vv;
      #pragma unroll
      for (int p = 0; p < 2; ++p) {
        int  gk  = k0 + ka + p;
        bool kOK = (gk < K);
        int  gkc = kOK ? gk : 0;
        {
          bool cOK = kOK && (row < Cout);
          float v  = Wg[coClamp0 + gkc];
          a0.h[2 * vv + p] = f2bf(cOK ? v : 0.f);
        }
        if (twoCo) {
          bool cOK = kOK && (16 + row < Cout);
          float v  = Wg[coClamp1 + gkc];
          a1.h[2 * vv + p] = f2bf(cOK ? v : 0.f);
        }
      }
    }

    acc0 = __builtin_amdgcn_wmma_f32_16x16x32_bf16(
        false, a0.v, false, b.v, (short)0, acc0, false, false);
    if (twoCo) {
      acc1 = __builtin_amdgcn_wmma_f32_16x16x32_bf16(
          false, a1.v, false, b.v, (short)0, acc1, false, false);
    }
  }

  // ---- epilogue: bias + optional relu, guarded stores (32-bit offsets) ----
  if (voxOK) {
    #pragma unroll
    for (int r = 0; r < 8; ++r) {
      int co = r + 8 * half;
      if (co < Cout) {
        float o = acc0[r] + Bi[co];
        if (relu) o = fmaxf(o, 0.f);
        Y[co * V + vox] = o;
      }
      if (twoCo) {
        int co2 = 16 + r + 8 * half;
        if (co2 < Cout) {
          float o = acc1[r] + Bi[co2];
          if (relu) o = fmaxf(o, 0.f);
          Y[co2 * V + vox] = o;
        }
      }
    }
  }
}

// ---------------------------------------------------------------------------
// Instance norm (per batch*channel plane) fused with PReLU.
// ---------------------------------------------------------------------------
__global__ void __launch_bounds__(256)
inorm_prelu_kernel(float* __restrict__ data, const float* __restrict__ alpha, int V)
{
  float* p = data + (size_t)blockIdx.x * V;
  __shared__ float s1[256], s2[256];
  __shared__ float sMu, sR;
  float sum = 0.f, ssq = 0.f;
  for (int i = threadIdx.x; i < V; i += 256) { float v = p[i]; sum += v; ssq += v * v; }
  s1[threadIdx.x] = sum; s2[threadIdx.x] = ssq;
  __syncthreads();
  for (int st = 128; st > 0; st >>= 1) {
    if ((int)threadIdx.x < st) { s1[threadIdx.x] += s1[threadIdx.x + st];
                                 s2[threadIdx.x] += s2[threadIdx.x + st]; }
    __syncthreads();
  }
  if (threadIdx.x == 0) {
    float m  = s1[0] / (float)V;
    float vr = fmaxf(s2[0] / (float)V - m * m, 0.f);
    sMu = m; sR = rsqrtf(vr + 1e-5f);
  }
  __syncthreads();
  float a = *alpha;
  float mu = sMu, rs = sR;
  for (int i = threadIdx.x; i < V; i += 256) {
    float v = (p[i] - mu) * rs;
    p[i] = v >= 0.f ? v : a * v;
  }
}

// ---------------------------------------------------------------------------
// Trilinear sampling helpers (border validity = zero outside, as in reference)
// ---------------------------------------------------------------------------
__device__ __forceinline__ void tri_setup(float z, float y, float x,
                                          int D, int H, int W,
                                          int idx[8], float wt[8])
{
  float zf = floorf(z), yf = floorf(y), xf = floorf(x);
  int z0 = (int)zf, y0 = (int)yf, x0 = (int)xf;
  float wz = z - zf, wy = y - yf, wx = x - xf;
  #pragma unroll
  for (int c = 0; c < 8; ++c) {
    int dz = c >> 2, dy = (c >> 1) & 1, dx = c & 1;
    int zi = z0 + dz, yi = y0 + dy, xi = x0 + dx;
    bool valid = (zi >= 0) && (zi < D) && (yi >= 0) && (yi < H) && (xi >= 0) && (xi < W);
    int zc = iclamp(zi, 0, D - 1), yc = iclamp(yi, 0, H - 1), xc = iclamp(xi, 0, W - 1);
    idx[c] = (zc * H + yc) * W + xc;
    float w = (dz ? wz : 1.f - wz) * (dy ? wy : 1.f - wy) * (dx ? wx : 1.f - wx);
    wt[c] = valid ? w : 0.f;
  }
}

// dst[c,p] = sample(src_c, p + flow(p)),  flow = (3,D,H,W) in (z,y,x) order
__global__ void __launch_bounds__(256)
warp_kernel(const float* __restrict__ src, const float* __restrict__ flow,
            float* __restrict__ dst, int C, int D, int H, int W)
{
  const int V = D * H * W;
  for (int p = blockIdx.x * blockDim.x + threadIdx.x; p < V;
       p += gridDim.x * blockDim.x) {
    int z = p / (H * W), rem = p % (H * W), y = rem / W, x = rem % W;
    int idx[8]; float wt[8];
    tri_setup((float)z + flow[p], (float)y + flow[V + p], (float)x + flow[2 * V + p],
              D, H, W, idx, wt);
    for (int c = 0; c < C; ++c) {
      const float* s = src + c * V;          // 32-bit offset
      float acc = 0.f;
      #pragma unroll
      for (int t = 0; t < 8; ++t) acc += s[idx[t]] * wt[t];
      dst[c * V + p] = acc;
    }
  }
}

// dst[c,p] = add[c,p] + sample(src_c, p + disp(p)),  C = 3
// covers vecint steps (src=disp=add=v) and composition (src=up, disp=add=flow)
__global__ void __launch_bounds__(256)
warp_add_kernel(const float* __restrict__ src, const float* __restrict__ disp,
                const float* __restrict__ add, float* __restrict__ dst,
                int D, int H, int W)
{
  const int V = D * H * W;
  for (int p = blockIdx.x * blockDim.x + threadIdx.x; p < V;
       p += gridDim.x * blockDim.x) {
    int z = p / (H * W), rem = p % (H * W), y = rem / W, x = rem % W;
    int idx[8]; float wt[8];
    tri_setup((float)z + disp[p], (float)y + disp[V + p], (float)x + disp[2 * V + p],
              D, H, W, idx, wt);
    #pragma unroll
    for (int c = 0; c < 3; ++c) {
      const float* s = src + c * V;
      float acc = add[c * V + p];
      #pragma unroll
      for (int t = 0; t < 8; ++t) acc += s[idx[t]] * wt[t];
      dst[c * V + p] = acc;
    }
  }
}

// ---------------------------------------------------------------------------
// Align-corners trilinear resize (separable-linear == trilinear) with fused
// output scale (used for the x2 flow upsample).
// ---------------------------------------------------------------------------
__device__ __forceinline__ void axc(int o, int O, int I, int& i0, float& t)
{
  if (O == I) { i0 = (o < I - 1) ? o : I - 2; t = (float)(o - i0); return; }
  float c = (float)o * ((float)(I - 1) / (float)(O - 1));
  int f = (int)floorf(c);
  i0 = iclamp(f, 0, I - 2);
  t = c - (float)i0;
}

__global__ void __launch_bounds__(256)
resize3_kernel(const float* __restrict__ src, float* __restrict__ dst, int C,
               int Di, int Hi, int Wi, int Do, int Ho, int Wo, float scale)
{
  const int Vo = Do * Ho * Wo, Vi = Di * Hi * Wi;
  const int n = C * Vo;
  for (int i = blockIdx.x * blockDim.x + threadIdx.x; i < n;
       i += gridDim.x * blockDim.x) {
    int c = i / Vo, p = i - c * Vo;
    int zo = p / (Ho * Wo), rem = p % (Ho * Wo), yo = rem / Wo, xo = rem % Wo;
    int z0, y0, x0; float tz, ty, tx;
    axc(zo, Do, Di, z0, tz);
    axc(yo, Ho, Hi, y0, ty);
    axc(xo, Wo, Wi, x0, tx);
    const float* s = src + c * Vi;
    float v = 0.f;
    #pragma unroll
    for (int t = 0; t < 8; ++t) {
      int dz = t >> 2, dy = (t >> 1) & 1, dx = t & 1;
      float w = (dz ? tz : 1.f - tz) * (dy ? ty : 1.f - ty) * (dx ? tx : 1.f - tx);
      v += s[((z0 + dz) * Hi + (y0 + dy)) * Wi + (x0 + dx)] * w;
    }
    dst[i] = v * scale;
  }
}

// out[c] = yi[c] + xw[c] (c<16);  out[16+c] = yi[c] - xw[c]
__global__ void __launch_bounds__(256)
makepair_kernel(const float* __restrict__ xw, const float* __restrict__ yi,
                float* __restrict__ out, int V)
{
  const int n = 16 * V;
  for (int i = blockIdx.x * blockDim.x + threadIdx.x; i < n;
       i += gridDim.x * blockDim.x) {
    float a = xw[i], b = yi[i];
    out[i] = b + a;
    out[i + n] = b - a;
  }
}

__global__ void __launch_bounds__(256)
scale_kernel(const float* __restrict__ in, float* __restrict__ out, float s, int n)
{
  for (int i = blockIdx.x * blockDim.x + threadIdx.x; i < n;
       i += gridDim.x * blockDim.x)
    out[i] = in[i] * s;
}

// ---------------------------------------------------------------------------
// Orchestration
// ---------------------------------------------------------------------------
extern "C" void kernel_launch(void* const* d_in, const int* in_sizes, int n_in,
                              void* d_out, int out_size, void* d_ws, size_t ws_size,
                              hipStream_t stream) {
  (void)in_sizes; (void)n_in; (void)out_size; (void)ws_size;

  const float* x   = (const float*)d_in[0];
  const float* yv  = (const float*)d_in[1];
  const float* ew1 = (const float*)d_in[2];
  const float* eb1 = (const float*)d_in[3];
  const float* ew2 = (const float*)d_in[4];
  const float* eb2 = (const float*)d_in[5];
  const float* ew3 = (const float*)d_in[6];
  const float* eb3 = (const float*)d_in[7];
  const float* pr1 = (const float*)d_in[8];
  const float* pr2 = (const float*)d_in[9];
  const float* pr3 = (const float*)d_in[10];
  const float* dw1 = (const float*)d_in[11];
  const float* db1 = (const float*)d_in[12];
  const float* dw2 = (const float*)d_in[13];
  const float* db2 = (const float*)d_in[14];
  const float* dw3 = (const float*)d_in[15];
  const float* db3 = (const float*)d_in[16];
  float* out = (float*)d_out;

  const int V0 = 64 * 64 * 64;
  int sdim[5], Vl[5];
  for (int l = 0; l < 5; ++l) { sdim[l] = 64 >> l; Vl[l] = sdim[l] * sdim[l] * sdim[l]; }

  // output layout: int flows L0..L4, then pos flows L0..L4
  size_t int_off[5], pos_off[5];
  int_off[0] = 0;
  for (int l = 1; l < 5; ++l) int_off[l] = int_off[l - 1] + (size_t)3 * Vl[l - 1];
  size_t total_int = int_off[4] + (size_t)3 * Vl[4];
  for (int l = 0; l < 5; ++l) pos_off[l] = total_int + int_off[l];

  // workspace bump allocator (floats)
  float* ws = (float*)d_ws;
  size_t off = 0;
  auto alloc = [&](size_t n) { float* p = ws + off; off += n; return p; };

  float* c1 = alloc((size_t)32 * V0);   // (2,16,64^3)
  float* c2 = alloc((size_t)64 * V0);   // (2,32,64^3)
  float* c3 = alloc((size_t)32 * V0);   // (2,16,64^3)  — stays live (level-0 features)
  float* px[5]; float* py[5];
  px[0] = c3; py[0] = c3 + (size_t)16 * V0;
  for (int l = 1; l < 5; ++l) { px[l] = alloc((size_t)16 * Vl[l]); py[l] = alloc((size_t)16 * Vl[l]); }
  float* vA = alloc((size_t)3 * V0);
  float* vB = alloc((size_t)3 * V0);
  float* upA = alloc((size_t)3 * V0);
  float* upB = alloc((size_t)3 * V0);

  // c1+c2 become a 96*V0 scratch pool after the encoder
  float* warped = c1;                          // 16*V0
  float* in2    = c1 + (size_t)16 * V0;        // 32*V0 (spans into c2)
  float* feat1  = c1 + (size_t)48 * V0;        // 32*V0
  float* feat2  = in2;                         // 27*V0, reuses in2 after feat1

  auto launch_conv = [&](const float* X, const float* Wg, const float* Bi, float* Y,
                         int Cin, int Cout, int s, int relu) {
    int V = s * s * s;
    int tiles = (V + 15) / 16;
    dim3 g((tiles + 7) / 8, 1, 1);       // 8 waves (8 tiles) per 256-thread block
    conv3d_wmma_kernel<<<g, 256, 0, stream>>>(X, Wg, Bi, Y, Cin, Cout, s, s, s, relu);
  };
  auto eblocks = [](long n) { return (unsigned)((n + 255) / 256); };

  // ---------------- encoder (batch = [x ; y]) ----------------
  launch_conv(x,  ew1, eb1, c1,                   1, 16, 64, 0);
  launch_conv(yv, ew1, eb1, c1 + (size_t)16 * V0, 1, 16, 64, 0);
  inorm_prelu_kernel<<<32, 256, 0, stream>>>(c1, pr1, V0);

  launch_conv(c1,                   ew2, eb2, c2,                   16, 32, 64, 0);
  launch_conv(c1 + (size_t)16 * V0, ew2, eb2, c2 + (size_t)32 * V0, 16, 32, 64, 0);
  inorm_prelu_kernel<<<64, 256, 0, stream>>>(c2, pr2, V0);

  launch_conv(c2,                   ew3, eb3, c3,                   32, 16, 64, 0);
  launch_conv(c2 + (size_t)32 * V0, ew3, eb3, c3 + (size_t)16 * V0, 32, 16, 64, 0);
  inorm_prelu_kernel<<<32, 256, 0, stream>>>(c3, pr3, V0);

  // ---------------- feature pyramid ----------------
  for (int l = 1; l < 5; ++l) {
    long n = 16L * Vl[l];
    resize3_kernel<<<eblocks(n), 256, 0, stream>>>(px[l - 1], px[l], 16,
        sdim[l - 1], sdim[l - 1], sdim[l - 1], sdim[l], sdim[l], sdim[l], 1.0f);
    resize3_kernel<<<eblocks(n), 256, 0, stream>>>(py[l - 1], py[l], 16,
        sdim[l - 1], sdim[l - 1], sdim[l - 1], sdim[l], sdim[l], sdim[l], 1.0f);
  }

  // ---------------- coarse-to-fine flow levels ----------------
  float* upCur = upA;
  float* upNext = upB;
  for (int l = 4; l >= 0; --l) {
    const int s = sdim[l], V = Vl[l];
    const bool haveUp = (l != 4);

    const float* xi = px[l];
    if (haveUp) {
      warp_kernel<<<eblocks(V), 256, 0, stream>>>(px[l], upCur, warped, 16, s, s, s);
      xi = warped;
    }
    makepair_kernel<<<eblocks(16L * V), 256, 0, stream>>>(xi, py[l], in2, V);

    launch_conv(in2,   dw1 + (size_t)l * 32 * 32 * 27, db1 + (size_t)l * 32, feat1, 32, 32, s, 1);
    launch_conv(feat1, dw2 + (size_t)l * 27 * 32 * 27, db2 + (size_t)l * 27, feat2, 32, 27, s, 1);

    float* preint = out + int_off[l];
    launch_conv(feat2, dw3 + (size_t)l * 3 * 27 * 27,  db3 + (size_t)l * 3,  preint, 27, 3, s, 0);

    // vecint: v = flow / 2^7; 7x: v = v + sample(v, p+v)
    scale_kernel<<<eblocks(3L * V), 256, 0, stream>>>(preint, vA, 1.0f / 128.0f, 3 * V);
    float* cur = vA; float* nxt = vB;
    for (int it = 0; it < 7; ++it) {
      warp_add_kernel<<<eblocks(V), 256, 0, stream>>>(cur, cur, cur, nxt, s, s, s);
      float* t = cur; cur = nxt; nxt = t;
    }

    float* posDst = out + pos_off[l];
    if (haveUp) {
      // pos = flow + sample(up_flow, p + flow)
      warp_add_kernel<<<eblocks(V), 256, 0, stream>>>(upCur, cur, cur, posDst, s, s, s);
    } else {
      hipMemcpyAsync(posDst, cur, sizeof(float) * 3 * V, hipMemcpyDeviceToDevice, stream);
    }

    if (l > 0) {
      const int so = 2 * s;
      resize3_kernel<<<eblocks(3L * so * so * so), 256, 0, stream>>>(
          posDst, upNext, 3, s, s, s, so, so, so, 2.0f);
      float* t = upCur; upCur = upNext; upNext = t;
    }
  }
}
